// FixedNet_63531156242916
// MI455X (gfx1250) — compile-verified, matrix operand order checked by reading, not run on hardware
//
#include <hip/hip_runtime.h>

// ---------------------------------------------------------------------------
// SNN forward (snntorch Leaky, subtract-reset) for MI455X / gfx1250, wave32.
//   T=100, B=256, DIN=768, H=1024, OUT=10
// 16 workgroups (one 16-row batch tile each, owns all layers & all steps),
// 256 threads = 8 waves, 8 WMMA N-tiles per wave.
// bf16 split-precision GEMMs on v_wmma_f32_16x16x32_bf16.
// ---------------------------------------------------------------------------

typedef __attribute__((ext_vector_type(8)))  float  v8f;
typedef __attribute__((ext_vector_type(16))) __bf16 v16bf;
typedef __attribute__((ext_vector_type(8)))  __bf16 v8bf;
typedef __attribute__((ext_vector_type(4)))  __bf16 v4bf;

#define T_STEPS 100
#define BATCH   256
#define DIN     768
#define HID     1024
#define NOUT    10
#define BETA    0.95f

// LDS strides (bf16 elements). 1552B / 2064B rows -> 4-bank skew so the two
// ds_load_b128 per A-fragment are conflict-free within each half-wave phase.
#define X_STRIDE   776
#define SPK_STRIDE 1032
#define SMEM_ELEMS 24832   // max(2*16*776, 16*1032) bf16 = 49,664 bytes

// d_ws bf16 layout (elements)
#define W1H_OFF 0u
#define W1L_OFF 786432u
#define W2H_OFF 1572864u
#define W2L_OFF 2621440u
#define W3H_OFF 3670016u
#define W3L_OFF 4718592u
#define W4H_OFF 5767168u
#define W4L_OFF 5777408u

#define SZH ((size_t)T_STEPS * BATCH * HID)   // 26,214,400
#define SZO ((size_t)T_STEPS * BATCH * NOUT)  // 256,000

// ---------------------------------------------------------------------------
// Split fp32 -> (bf16 hi, bf16 lo) so hi+lo reconstructs ~fp32.
// ---------------------------------------------------------------------------
__global__ void split_bf16_kernel(const float* __restrict__ src,
                                  __bf16* __restrict__ hi,
                                  __bf16* __restrict__ lo, int n) {
  int stride = gridDim.x * blockDim.x;
  for (int i = blockIdx.x * blockDim.x + threadIdx.x; i < n; i += stride) {
    float f = src[i];
    __bf16 h = (__bf16)f;
    hi[i] = h;
    lo[i] = (__bf16)(f - (float)h);
  }
}

// ---------------------------------------------------------------------------
// WMMA helpers
// ---------------------------------------------------------------------------
static __device__ __forceinline__ v8f wmma_bf16(v16bf a, v16bf b, v8f c) {
  return __builtin_amdgcn_wmma_f32_16x16x32_bf16(
      /*neg_a=*/false, a, /*neg_b=*/false, b,
      /*c_mod=*/(short)0, c, /*reuse_a=*/false, /*reuse_b=*/false);
}

// A/B fragment for 16x16x32 bf16 WMMA. Per lane (row/col = lane&15,
// h2 = lane>>4): K-runs [k0+8*h2 .. +7] and [k0+16+8*h2 .. +7].
static __device__ __forceinline__ v16bf load_frag(const __bf16* rowPtr, int k0, int h2) {
  const v8bf a0 = *(const v8bf*)(rowPtr + k0 + 8 * h2);
  const v8bf a1 = *(const v8bf*)(rowPtr + k0 + 16 + 8 * h2);
  v16bf r;
#pragma unroll
  for (int i = 0; i < 8; ++i) { r[i] = a0[i]; r[i + 8] = a1[i]; }
  return r;
}

// ---------------------------------------------------------------------------
// One hidden layer (H=1024 outputs): GEMM sweep + LIF + spike store to LDS.
// SPLIT_A: also use the lo part of the A operand (layer 1, fp32 inputs).
// ---------------------------------------------------------------------------
template <bool SPLIT_A>
static __device__ __forceinline__ void run_layer(
    const __bf16* __restrict__ Wh, const __bf16* __restrict__ Wl,
    const float* __restrict__ bias,
    float* __restrict__ curBase, float* __restrict__ memBase,
    int K, const __bf16* aHi, const __bf16* aLo, int aStride,
    __bf16* spkOut, int t, int b0) {
  const int tid  = threadIdx.x;
  const int lane = tid & 31;
  const int wave = tid >> 5;
  const int m16  = lane & 15;
  const int h2   = lane >> 4;

  v8f acc[8];
#pragma unroll
  for (int i = 0; i < 8; ++i) {
    float bv = bias[(wave * 8 + i) * 16 + m16];
#pragma unroll
    for (int r = 0; r < 8; ++r) acc[i][r] = bv;
  }

  const __bf16* aHiRow = aHi + m16 * aStride;
  const __bf16* aLoRow = aLo + m16 * aStride;

  for (int k0 = 0; k0 < K; k0 += 32) {
    v16bf Ah = load_frag(aHiRow, k0, h2);
    v16bf Al;
    if (SPLIT_A) Al = load_frag(aLoRow, k0, h2);
#pragma unroll
    for (int i = 0; i < 8; ++i) {
      int col = (wave * 8 + i) * 16 + m16;
      v16bf Bh = load_frag(Wh + (size_t)col * K, k0, h2);
      v16bf Bl = load_frag(Wl + (size_t)col * K, k0, h2);
      acc[i] = wmma_bf16(Ah, Bh, acc[i]);
      acc[i] = wmma_bf16(Ah, Bl, acc[i]);
      if (SPLIT_A) acc[i] = wmma_bf16(Al, Bh, acc[i]);
    }
  }

  __syncthreads();  // all waves done reading A spikes from LDS

  // LIF: mem = beta*mem_prev + cur - (mem_prev>1), spk = (mem-1 > 0)
#pragma unroll
  for (int i = 0; i < 8; ++i) {
    int col = (wave * 8 + i) * 16 + m16;
#pragma unroll
    for (int r = 0; r < 8; ++r) {
      int mrow = r + 8 * h2;  // C-layout: VGPR r holds rows r / r+8
      size_t o = ((size_t)t * BATCH + (b0 + mrow)) * HID + col;
      float cur = acc[i][r];
      float mp = 0.0f;
      if (t > 0) mp = memBase[o - (size_t)BATCH * HID];
      float reset = (mp > 1.0f) ? 1.0f : 0.0f;
      float mn = BETA * mp + cur - reset;
      __builtin_nontemporal_store(cur, curBase + o);  // pure stream, keep out of L2
      memBase[o] = mn;                                // re-read next step (L2 hit)
      spkOut[mrow * SPK_STRIDE + col] = (mn > 1.0f) ? (__bf16)1.0f : (__bf16)0.0f;
    }
  }
  __syncthreads();  // spikes visible to next layer
}

// ---------------------------------------------------------------------------
// Output layer (OUT=10): computed redundantly by all 8 waves (keeps EXEC
// all-ones around WMMA); wave 0 does LIF + stores. Rows >= 10 clamp to row 0.
// ---------------------------------------------------------------------------
static __device__ __forceinline__ void run_layer4(
    const __bf16* __restrict__ W4h, const __bf16* __restrict__ W4l,
    const float* __restrict__ b4,
    float* __restrict__ cur4, float* __restrict__ mem4,
    const __bf16* spk, int t, int b0) {
  const int tid  = threadIdx.x;
  const int lane = tid & 31;
  const int m16  = lane & 15;
  const int h2   = lane >> 4;

  v8f acc;
#pragma unroll
  for (int r = 0; r < 8; ++r) acc[r] = 0.0f;

  const int jr = (m16 < NOUT) ? m16 : 0;  // avoid OOB weight reads
  const __bf16* wph = W4h + (size_t)jr * HID;
  const __bf16* wpl = W4l + (size_t)jr * HID;
  const __bf16* aRow = spk + m16 * SPK_STRIDE;

  for (int k0 = 0; k0 < HID; k0 += 32) {
    v16bf A  = load_frag(aRow, k0, h2);
    v16bf Bh = load_frag(wph, k0, h2);
    v16bf Bl = load_frag(wpl, k0, h2);
    acc = wmma_bf16(A, Bh, acc);
    acc = wmma_bf16(A, Bl, acc);
  }

  if (tid < 32 && m16 < NOUT) {
    float bv = b4[m16];
#pragma unroll
    for (int r = 0; r < 8; ++r) {
      int mrow = r + 8 * h2;
      size_t o = ((size_t)t * BATCH + (b0 + mrow)) * NOUT + m16;
      float cur = acc[r] + bv;
      float mp = 0.0f;
      if (t > 0) mp = mem4[o - (size_t)BATCH * NOUT];
      float reset = (mp > 1.0f) ? 1.0f : 0.0f;
      float mn = BETA * mp + cur - reset;
      __builtin_nontemporal_store(cur, cur4 + o);
      mem4[o] = mn;
    }
  }
  // no barrier here: loop-top barrier in caller fences LDS reuse
}

// ---------------------------------------------------------------------------
// Main persistent kernel: 16 blocks x 256 threads; block owns batch rows
// [b0, b0+16) across all 4 layers and all 100 time steps.
// ---------------------------------------------------------------------------
__global__ __launch_bounds__(256)
void snn_forward_kernel(const float* __restrict__ x,
                        const float* __restrict__ b1, const float* __restrict__ b2,
                        const float* __restrict__ b3, const float* __restrict__ b4,
                        const __bf16* __restrict__ ws,
                        float* __restrict__ out) {
  __shared__ __bf16 smem[SMEM_ELEMS];  // 49,664 bytes

  const int tid = threadIdx.x;
  const int b0  = blockIdx.x * 16;

  const __bf16* W1h = ws + W1H_OFF;  const __bf16* W1l = ws + W1L_OFF;
  const __bf16* W2h = ws + W2H_OFF;  const __bf16* W2l = ws + W2L_OFF;
  const __bf16* W3h = ws + W3H_OFF;  const __bf16* W3l = ws + W3L_OFF;
  const __bf16* W4h = ws + W4H_OFF;  const __bf16* W4l = ws + W4L_OFF;

  float* cur1 = out;
  float* cur2 = out + SZH;
  float* cur3 = out + 2 * SZH;
  float* cur4 = out + 3 * SZH;
  float* m1   = out + 3 * SZH + SZO;
  float* m2   = m1 + SZH;
  float* m3   = m1 + 2 * SZH;
  float* m4   = m1 + 3 * SZH;

  __bf16* xh = smem;                 // layer-1 A (hi), 16 x X_STRIDE
  __bf16* xl = smem + 16 * X_STRIDE; // layer-1 A (lo)
  __bf16* sp = smem;                 // spike buffer (reused in place)

  for (int t = 0; t < T_STEPS; ++t) {
    __syncthreads();  // fences prior-step spk3 LDS reads before restaging

    // Stage x[:, t, :] tile -> LDS as bf16 hi/lo. 3072 float4 / 256 threads.
#pragma unroll
    for (int it = 0; it < 12; ++it) {
      int lin = tid + it * 256;        // float4 index
      int m = lin / 192;
      int c = (lin % 192) * 4;
      const float4 v = *(const float4*)(x + ((size_t)(b0 + m) * T_STEPS + t) * DIN + c);
      v4bf hv, lv;
      hv[0] = (__bf16)v.x; lv[0] = (__bf16)(v.x - (float)hv[0]);
      hv[1] = (__bf16)v.y; lv[1] = (__bf16)(v.y - (float)hv[1]);
      hv[2] = (__bf16)v.z; lv[2] = (__bf16)(v.z - (float)hv[2]);
      hv[3] = (__bf16)v.w; lv[3] = (__bf16)(v.w - (float)hv[3]);
      *(v4bf*)(xh + m * X_STRIDE + c) = hv;
      *(v4bf*)(xl + m * X_STRIDE + c) = lv;
    }
    __syncthreads();

    // Layer 1: A = x (hi+lo split), 3-term WMMA
    run_layer<true >(W1h, W1l, b1, cur1, m1, DIN, xh, xl, X_STRIDE, sp, t, b0);
    // Layers 2/3: A = spikes (exact in bf16), 2-term WMMA
    run_layer<false>(W2h, W2l, b2, cur2, m2, HID, sp, sp, SPK_STRIDE, sp, t, b0);
    run_layer<false>(W3h, W3l, b3, cur3, m3, HID, sp, sp, SPK_STRIDE, sp, t, b0);
    // Layer 4
    run_layer4(W4h, W4l, b4, cur4, m4, sp, t, b0);
  }
}

// ---------------------------------------------------------------------------
extern "C" void kernel_launch(void* const* d_in, const int* in_sizes, int n_in,
                              void* d_out, int out_size, void* d_ws, size_t ws_size,
                              hipStream_t stream) {
  const float* x  = (const float*)d_in[0];
  const float* W1 = (const float*)d_in[1];
  const float* b1 = (const float*)d_in[2];
  const float* W2 = (const float*)d_in[3];
  const float* b2 = (const float*)d_in[4];
  const float* W3 = (const float*)d_in[5];
  const float* b3 = (const float*)d_in[6];
  const float* W4 = (const float*)d_in[7];
  const float* b4 = (const float*)d_in[8];

  __bf16* ws = (__bf16*)d_ws;  // 5,787,648 bf16 = 11.6 MB

  split_bf16_kernel<<<768, 256, 0, stream>>>(W1, ws + W1H_OFF, ws + W1L_OFF, DIN * HID);
  split_bf16_kernel<<<1024, 256, 0, stream>>>(W2, ws + W2H_OFF, ws + W2L_OFF, HID * HID);
  split_bf16_kernel<<<1024, 256, 0, stream>>>(W3, ws + W3H_OFF, ws + W3L_OFF, HID * HID);
  split_bf16_kernel<<<40, 256, 0, stream>>>(W4, ws + W4H_OFF, ws + W4L_OFF, NOUT * HID);

  snn_forward_kernel<<<16, 256, 0, stream>>>(x, b1, b2, b3, b4, ws, (float*)d_out);
}
